// VarRnnCellWrapper_1597727834467
// MI455X (gfx1250) — matrix operand order repaired and checked by reading, not compiled, not run on hardware
//
#include <hip/hip_runtime.h>
#include <cstdint>

#define T_STEPS 512
#define BATCH   256
#define DIM     256
#define HID     256

typedef __attribute__((ext_vector_type(16))) __bf16 v16bf;
typedef __attribute__((ext_vector_type(8)))  float  v8f;

union V16BF { uint32_t u[8]; v16bf v; };

__device__ __forceinline__ uint32_t pack_bf16(float a, float b) {
    uint32_t ua = __builtin_bit_cast(uint32_t, a);
    uint32_t ub = __builtin_bit_cast(uint32_t, b);
    ua += 0x7FFFu + ((ua >> 16) & 1u);      // round-to-nearest-even
    ub += 0x7FFFu + ((ub >> 16) & 1u);
    return (ua >> 16) | (ub & 0xFFFF0000u);
}

__device__ __forceinline__ float sigmoid_f(float x) {
    return 1.0f / (1.0f + __expf(-x));
}

__device__ __forceinline__ float tanh_f(float x) {
    x = fminf(fmaxf(x, -15.0f), 15.0f);
    float e = __expf(2.0f * x);
    return (e - 1.0f) / (e + 1.0f);
}

__device__ __forceinline__ void load_a16(const uint32_t* __restrict__ p, int base, V16BF& a) {
    uint4 q0 = *(const uint4*)&p[base];
    uint4 q1 = *(const uint4*)&p[base + 4];
    a.u[0] = q0.x; a.u[1] = q0.y; a.u[2] = q0.z; a.u[3] = q0.w;
    a.u[4] = q1.x; a.u[5] = q1.y; a.u[6] = q1.z; a.u[7] = q1.w;
}

// One-time f32 -> packed bf16 conversion of W_ih and W_hh into workspace.
__global__ void convert_w_kernel(const float* __restrict__ Wih,
                                 const float* __restrict__ Whh,
                                 uint32_t* __restrict__ wih_bf,
                                 uint32_t* __restrict__ whh_bf) {
    int idx = blockIdx.x * blockDim.x + threadIdx.x;   // dword index
    const int ndw = (4 * HID * DIM) / 2;               // 131072
    if (idx < ndw) {
        wih_bf[idx] = pack_bf16(Wih[2 * idx], Wih[2 * idx + 1]);
        whh_bf[idx] = pack_bf16(Whh[2 * idx], Whh[2 * idx + 1]);
    }
}

// One LSTM time step.
// Grid: (4 col-groups, 8 row-pairs). Block: 128 threads = 4 waves.
// Each block owns 32 batch rows (two 16-row WMMA sub-blocks, sharing B tiles).
// Wave w owns h-column tile ct = cg*4 + w: gate tiles at columns ct*16 + g*HID.
__global__ __launch_bounds__(128)
void lstm_step_kernel(const float* __restrict__ x_t,        // [B, D] slice for step t
                      const float* __restrict__ h_in,       // [B, H] prev h state (h0 at t=0)
                      const float* __restrict__ c_in,       // [B, H] prev c state (c0 at t=0)
                      const float* __restrict__ mask_x,     // [B, D]
                      const float* __restrict__ mask_h,     // [B, H]
                      const float* __restrict__ b_ih,       // [4H]
                      const float* __restrict__ b_hh,       // [4H]
                      const int*   __restrict__ batch_sizes,// [T]
                      const uint32_t* __restrict__ wih_bf,  // [4H][D/2] packed bf16
                      const uint32_t* __restrict__ whh_bf,  // [4H][H/2] packed bf16
                      float* __restrict__ h_out,            // [B, H] next h state
                      float* __restrict__ c_out,            // [B, H] (== d_out cn region)
                      float* __restrict__ out,              // [T, B, H] (d_out base)
                      float* __restrict__ hn_region,        // [B, H]
                      int t) {
    __shared__ __align__(16) uint32_t lds_xi[32 * 128];     // xi bf16 pairs, 32 rows x 256 k
    __shared__ __align__(16) uint32_t lds_hi[32 * 128];     // hi bf16 pairs

    const int tid   = threadIdx.x;
    const int rbase = blockIdx.y * 32;  // first batch row of this block
    const int cg    = blockIdx.x;       // col group 0..3
    const int lane  = tid & 31;
    const int wv    = tid >> 5;         // wave 0..3
    const int ct    = cg * 4 + wv;      // h-col tile 0..15
    const int c2    = lane >> 4;        // lane half 0/1
    const int ln    = lane & 15;

    // ---- Stage dropout-masked bf16 A-operands (xi, hi) into LDS ----
    #pragma unroll
    for (int i = tid; i < 32 * 128; i += 128) {
        int row  = i >> 7;              // 0..31
        int dcol = i & 127;             // dword column
        int b    = rbase + row;
        int k0   = dcol * 2;
        float xv0 = x_t[b * DIM + k0]     * mask_x[b * DIM + k0];
        float xv1 = x_t[b * DIM + k0 + 1] * mask_x[b * DIM + k0 + 1];
        lds_xi[i] = pack_bf16(xv0, xv1);
        float hv0 = h_in[b * HID + k0]     * mask_h[b * HID + k0];
        float hv1 = h_in[b * HID + k0 + 1] * mask_h[b * HID + k0 + 1];
        lds_hi[i] = pack_bf16(hv0, hv1);
    }
    __syncthreads();

    // ---- GEMM: 2 row sub-blocks x 4 gate tiles; K = 256 in 8 chunks of 32.
    //      Each B tile (weights) is reused by both row sub-blocks.
    v8f acc[2][4] = {{v8f{}, v8f{}, v8f{}, v8f{}}, {v8f{}, v8f{}, v8f{}, v8f{}}};

    #pragma unroll
    for (int chunk = 0; chunk < 8; ++chunk) {
        // A operands: lane ln holds row ln (sub-block s adds 16 rows);
        // K contiguous {0..15} for lanes 0-15, {16..31} for 16-31.
        V16BF ax0, ax1, ah0, ah1;
        {
            int a0 = (ln << 7) + (chunk << 4) + (c2 << 3);
            int a1 = a0 + (16 << 7);
            load_a16(lds_xi, a0, ax0);
            load_a16(lds_xi, a1, ax1);
            load_a16(lds_hi, a0, ah0);
            load_a16(lds_hi, a1, ah1);
        }
        #pragma unroll
        for (int g = 0; g < 4; ++g) {
            // B operand: lane (16*c2 + ln) holds column ln, K half c2.
            int j = g * HID + ct * 16 + ln;                 // weight row (gate output col)
            size_t wb = (size_t)j * (DIM / 2) + (chunk << 4) + (c2 << 3);
            V16BF bi, bh;
            {
                uint4 q0 = *(const uint4*)&wih_bf[wb];
                uint4 q1 = *(const uint4*)&wih_bf[wb + 4];
                bi.u[0] = q0.x; bi.u[1] = q0.y; bi.u[2] = q0.z; bi.u[3] = q0.w;
                bi.u[4] = q1.x; bi.u[5] = q1.y; bi.u[6] = q1.z; bi.u[7] = q1.w;
                uint4 r0 = *(const uint4*)&whh_bf[wb];
                uint4 r1 = *(const uint4*)&whh_bf[wb + 4];
                bh.u[0] = r0.x; bh.u[1] = r0.y; bh.u[2] = r0.z; bh.u[3] = r0.w;
                bh.u[4] = r1.x; bh.u[5] = r1.y; bh.u[6] = r1.z; bh.u[7] = r1.w;
            }
            acc[0][g] = __builtin_amdgcn_wmma_f32_16x16x32_bf16(
                false, ax0.v, false, bi.v, (short)0, acc[0][g], false, false);
            acc[1][g] = __builtin_amdgcn_wmma_f32_16x16x32_bf16(
                false, ax1.v, false, bi.v, (short)0, acc[1][g], false, false);
            acc[0][g] = __builtin_amdgcn_wmma_f32_16x16x32_bf16(
                false, ah0.v, false, bh.v, (short)0, acc[0][g], false, false);
            acc[1][g] = __builtin_amdgcn_wmma_f32_16x16x32_bf16(
                false, ah1.v, false, bh.v, (short)0, acc[1][g], false, false);
        }
    }

    // ---- Biases (depend only on output column = ln within gate tile) ----
    float bias[4];
    #pragma unroll
    for (int g = 0; g < 4; ++g) {
        int j = g * HID + ct * 16 + ln;
        bias[g] = b_ih[j] + b_hh[j];
    }

    const int bs  = batch_sizes[t];
    const int col = ct * 16 + ln;

    // ---- Elementwise LSTM cell update; C/D layout: M = r + 8*c2, N = ln ----
    #pragma unroll
    for (int s = 0; s < 2; ++s) {
        #pragma unroll
        for (int r = 0; r < 8; ++r) {
            int brow = rbase + s * 16 + r + c2 * 8;
            float gi = acc[s][0][r] + bias[0];
            float gf = acc[s][1][r] + bias[1];
            float gg = acc[s][2][r] + bias[2];
            float go = acc[s][3][r] + bias[3];

            float si = sigmoid_f(gi);
            float sf = sigmoid_f(gf);
            float so = sigmoid_f(go);
            float tg = tanh_f(gg);

            float c_prev = c_in[brow * HID + col];
            float h_prev = h_in[brow * HID + col];

            float c_new = sf * c_prev + si * tg;
            float h_new = so * tanh_f(c_new);

            bool  act    = brow < bs;
            float c_keep = act ? c_new : c_prev;
            float h_keep = act ? h_new : h_prev;

            c_out[brow * HID + col] = c_keep;
            h_out[brow * HID + col] = h_keep;
            out[((size_t)t * BATCH + brow) * HID + col] = act ? h_new : 0.0f;
            if (t == T_STEPS - 1) {
                hn_region[brow * HID + col] = h_keep;
            }
        }
    }
}

extern "C" void kernel_launch(void* const* d_in, const int* in_sizes, int n_in,
                              void* d_out, int out_size, void* d_ws, size_t ws_size,
                              hipStream_t stream) {
    (void)in_sizes; (void)n_in; (void)out_size; (void)ws_size;

    const float* x      = (const float*)d_in[0];   // [T, B, D]
    const float* h0     = (const float*)d_in[1];   // [B, H]
    const float* c0     = (const float*)d_in[2];   // [B, H]
    const float* mask_x = (const float*)d_in[3];   // [B, D]
    const float* mask_h = (const float*)d_in[4];   // [B, H]
    const float* W_ih   = (const float*)d_in[5];   // [4H, D]
    const float* W_hh   = (const float*)d_in[6];   // [4H, H]
    const float* b_ih   = (const float*)d_in[7];   // [4H]
    const float* b_hh   = (const float*)d_in[8];   // [4H]
    const int* batch_sizes = (const int*)d_in[9];  // [T]

    float* out       = (float*)d_out;                         // [T,B,H]
    float* hn_region = out + (size_t)T_STEPS * BATCH * HID;   // [B,H]
    float* cn_region = hn_region + BATCH * HID;               // [B,H]

    // Workspace: [wih_bf 512KB][whh_bf 512KB][hbuf0 256KB][hbuf1 256KB]
    uint32_t* ws      = (uint32_t*)d_ws;
    uint32_t* wih_bf  = ws;
    uint32_t* whh_bf  = ws + (4 * HID * DIM) / 2;
    float*    hbuf0   = (float*)(ws + 2 * ((4 * HID * DIM) / 2));
    float*    hbuf1   = hbuf0 + BATCH * HID;

    {
        const int ndw = (4 * HID * DIM) / 2;
        convert_w_kernel<<<dim3((ndw + 255) / 256), dim3(256), 0, stream>>>(
            W_ih, W_hh, wih_bf, whh_bf);
    }

    for (int t = 0; t < T_STEPS; ++t) {
        const float* h_in = (t == 0) ? h0 : ((t & 1) ? hbuf1 : hbuf0);
        const float* c_in = (t == 0) ? c0 : cn_region;
        float* h_out = ((t + 1) & 1) ? hbuf1 : hbuf0;
        lstm_step_kernel<<<dim3(4, 8), dim3(128), 0, stream>>>(
            x + (size_t)t * BATCH * DIM, h_in, c_in, mask_x, mask_h,
            b_ih, b_hh, batch_sizes, wih_bf, whh_bf,
            h_out, cn_region, out, hn_region, t);
    }
}
